// MultiHeadAttention_61314953117930
// MI455X (gfx1250) — matrix operand order repaired and checked by reading, not compiled, not run on hardware
//
#include <hip/hip_runtime.h>

// ---------------------------------------------------------------------------
// MI455X (gfx1250) GQA attention layer.
// Compute-bound (~40 GFLOP vs ~50 MB of traffic; fits in 192MB L2), so the
// whole pipeline is driven through V_WMMA_F32_16X16X32_BF16 (f32 accumulate).
// wave32 everywhere; WMMA fragment layouts per CDNA5 ISA 7.12.2.
// ---------------------------------------------------------------------------

typedef __attribute__((ext_vector_type(16))) __bf16 v16bf;
typedef __attribute__((ext_vector_type(8)))  float  v8f;

union ABFrag { uint4 u[2]; v16bf v; };

#define S_LEN   2048
#define D_MODEL 1024
#define NQKV    1536      // 1024 q + 256 k + 256 v
#define N_H     16
#define N_KVH   4
#define HD      64
#define SCALE_ATT 0.125f  // 1/sqrt(64)
#define NEG_MASK (-1.0e9f)

__device__ __forceinline__ unsigned short f2bf(float f) {
    unsigned int u = __float_as_uint(f);
    u += 0x7FFFu + ((u >> 16) & 1u);          // round-to-nearest-even
    return (unsigned short)(u >> 16);
}

__device__ __forceinline__ float rowmax16(float x) {
#pragma unroll
    for (int m = 8; m >= 1; m >>= 1) x = fmaxf(x, __shfl_xor(x, m, 32));
    return x;
}
__device__ __forceinline__ float rowsum16(float x) {
#pragma unroll
    for (int m = 8; m >= 1; m >>= 1) x += __shfl_xor(x, m, 32);
    return x;
}

// ---------------------------------------------------------------------------
// Prep kernels: fp32 -> bf16 conversion and weight transposition so that all
// WMMA B-fragments become contiguous-K loads (lane = output col, K contiguous)
// ---------------------------------------------------------------------------
__global__ void cvt_bf16_kernel(const float* __restrict__ src,
                                unsigned short* __restrict__ dst, int n) {
    int i = blockIdx.x * blockDim.x + threadIdx.x;
    if (i < n) dst[i] = f2bf(src[i]);
}

__global__ void pack_wqkv_kernel(const float* __restrict__ Wq,
                                 const float* __restrict__ Wk,
                                 const float* __restrict__ Wv,
                                 unsigned short* __restrict__ Wt) {
    int idx = blockIdx.x * blockDim.x + threadIdx.x;   // Wt[n][k], n<1536,k<1024
    if (idx >= NQKV * D_MODEL) return;
    int n = idx >> 10, k = idx & 1023;
    float v;
    if (n < 1024)      v = Wq[(size_t)k * 1024 + n];
    else if (n < 1280) v = Wk[(size_t)k * 256 + (n - 1024)];
    else               v = Wv[(size_t)k * 256 + (n - 1280)];
    Wt[idx] = f2bf(v);
}

__global__ void transpose_wo_kernel(const float* __restrict__ Wo,
                                    unsigned short* __restrict__ Wt) {
    int idx = blockIdx.x * blockDim.x + threadIdx.x;   // Wt[n][k]
    if (idx >= D_MODEL * D_MODEL) return;
    int n = idx >> 10, k = idx & 1023;
    Wt[idx] = f2bf(Wo[(size_t)k * 1024 + n]);
}

// ---------------------------------------------------------------------------
// Kernel 1: fused QKV projection  (4096 x 1536 = hs @ [Wq|Wk|Wv])
//   128x128 block tile, K-step 32, 8 waves (2x4), each wave 64x32 (4x2 frags).
//   Epilogue applies RoPE to q/k (pairwise rotation via shfl_xor lane swap)
//   and scatters q:[b][h][s][d], k:[b][g][s][d], v transposed:[b][g][d][s].
// ---------------------------------------------------------------------------
__global__ __launch_bounds__(256)
void qkv_gemm_kernel(const unsigned short* __restrict__ Ag,   // bf16 hs (4096x1024)
                     const unsigned short* __restrict__ Bg,   // bf16 WqkvT (1536x1024)
                     const float* __restrict__ cosb,          // (B,S,32)
                     const float* __restrict__ sinb,
                     unsigned short* __restrict__ qo,
                     unsigned short* __restrict__ ko,
                     unsigned short* __restrict__ vto) {
    __shared__ alignas(16) unsigned short As[128 * 40];
    __shared__ alignas(16) unsigned short Bs[128 * 40];

    const int tid  = threadIdx.x;
    const int wid  = tid >> 5;
    const int lane = tid & 31;
    const int l16  = lane & 15;
    const int hi   = (lane >> 4) & 1;
    const int waveM = wid & 1;   // 0..1 -> 64 rows each
    const int waveN = wid >> 1;  // 0..3 -> 32 cols each
    const int m0 = blockIdx.y * 128;
    const int n0 = blockIdx.x * 128;

    v8f acc[4][2];
#pragma unroll
    for (int a = 0; a < 4; ++a)
#pragma unroll
        for (int b = 0; b < 2; ++b)
#pragma unroll
            for (int e = 0; e < 8; ++e) acc[a][b][e] = 0.0f;

    for (int k0 = 0; k0 < D_MODEL; k0 += 32) {
        __syncthreads();
        {   // stage 128x32 bf16 tiles of A and B (2 x uint4 per thread each)
            int j = tid;
#pragma unroll
            for (int it = 0; it < 2; ++it, j += 256) {
                int row = j >> 2, cc = (j & 3) << 3;
                *reinterpret_cast<uint4*>(&As[row * 40 + cc]) =
                    *reinterpret_cast<const uint4*>(&Ag[(size_t)(m0 + row) * D_MODEL + k0 + cc]);
                *reinterpret_cast<uint4*>(&Bs[row * 40 + cc]) =
                    *reinterpret_cast<const uint4*>(&Bg[(size_t)(n0 + row) * D_MODEL + k0 + cc]);
            }
        }
        if (k0 + 32 < D_MODEL) {   // global_prefetch_b8 hint for next K tile
            __builtin_prefetch(&Ag[(size_t)(m0 + (tid & 127)) * D_MODEL + k0 + 32], 0, 3);
            __builtin_prefetch(&Bg[(size_t)(n0 + (tid & 127)) * D_MODEL + k0 + 32], 0, 3);
        }
        __syncthreads();

        ABFrag a[4], b[2];
#pragma unroll
        for (int fm = 0; fm < 4; ++fm) {       // A frag: lane row l16, K chunks c / c+16
            int am = waveM * 64 + fm * 16 + l16;
            int c  = hi * 8;
            a[fm].u[0] = *reinterpret_cast<const uint4*>(&As[am * 40 + c]);
            a[fm].u[1] = *reinterpret_cast<const uint4*>(&As[am * 40 + c + 16]);
        }
#pragma unroll
        for (int fn = 0; fn < 2; ++fn) {       // B frag: lane col l16, K chunks c / c+16
            int bn = waveN * 32 + fn * 16 + l16;
            int c  = hi * 8;
            b[fn].u[0] = *reinterpret_cast<const uint4*>(&Bs[bn * 40 + c]);
            b[fn].u[1] = *reinterpret_cast<const uint4*>(&Bs[bn * 40 + c + 16]);
        }
#pragma unroll
        for (int fm = 0; fm < 4; ++fm)
#pragma unroll
            for (int fn = 0; fn < 2; ++fn)
                acc[fm][fn] = __builtin_amdgcn_wmma_f32_16x16x32_bf16(
                    false, a[fm].v, false, b[fn].v, (short)0, acc[fm][fn], false, false);
    }

    // epilogue: RoPE + scatter
#pragma unroll
    for (int fm = 0; fm < 4; ++fm) {
#pragma unroll
        for (int fn = 0; fn < 2; ++fn) {
            int n = n0 + waveN * 32 + fn * 16 + l16;
#pragma unroll
            for (int e = 0; e < 8; ++e) {
                int row  = m0 + waveM * 64 + fm * 16 + e + hi * 8;
                int bb   = row >> 11;           // / 2048
                int spos = row & 2047;
                float val = acc[fm][fn][e];
                float partner = __shfl_xor(val, 1, 32);   // rope pair (all lanes active)
                if (n < 1024) {
                    int h = n >> 6, d = n & 63, p = d >> 1;
                    size_t ci = ((size_t)bb * S_LEN + spos) * 32 + p;
                    float c = cosb[ci], s = sinb[ci];
                    float o = c * val + ((d & 1) ? s * partner : -s * partner);
                    qo[(((size_t)bb * N_H + h) * S_LEN + spos) * HD + d] = f2bf(o);
                } else if (n < 1280) {
                    int nn = n - 1024, g = nn >> 6, d = nn & 63, p = d >> 1;
                    size_t ci = ((size_t)bb * S_LEN + spos) * 32 + p;
                    float c = cosb[ci], s = sinb[ci];
                    float o = c * val + ((d & 1) ? s * partner : -s * partner);
                    ko[(((size_t)bb * N_KVH + g) * S_LEN + spos) * HD + d] = f2bf(o);
                } else {
                    int nn = n - 1280, g = nn >> 6, d = nn & 63;
                    vto[(((size_t)bb * N_KVH + g) * HD + d) * S_LEN + spos] = f2bf(val);
                }
            }
        }
    }
}

// ---------------------------------------------------------------------------
// Kernel 2: flash-style causal GQA attention. One wave per 16-query tile,
// 32-key blocks: 4 score WMMAs (K=64 head dim), online softmax (shfl row
// reductions), P re-layout via per-wave LDS tile, 4 PV WMMAs into f32 acc.
// ---------------------------------------------------------------------------
__global__ __launch_bounds__(256)
void attn_kernel(const unsigned short* __restrict__ qws,
                 const unsigned short* __restrict__ kws,
                 const unsigned short* __restrict__ vtws,
                 unsigned short* __restrict__ ctxws) {
    __shared__ alignas(16) unsigned short Ps[8][16 * 40];

    const int tid  = threadIdx.x;
    const int wid  = tid >> 5;
    const int lane = tid & 31;
    const int l16  = lane & 15;
    const int hi   = (lane >> 4) & 1;
    const int h = blockIdx.y;
    const int b = blockIdx.z;
    const int g = h >> 2;                        // GQA: kv head = h / 4
    const int qbase = (blockIdx.x * 8 + wid) * 16;

    const unsigned short* qp = qws  + ((size_t)b * N_H   + h) * S_LEN * HD;
    const unsigned short* kp = kws  + ((size_t)b * N_KVH + g) * S_LEN * HD;
    const unsigned short* vp = vtws + ((size_t)b * N_KVH + g) * HD * S_LEN;

    ABFrag qa[2];                                // q tile 16 x 64 as two K=32 A-frags
#pragma unroll
    for (int t = 0; t < 2; ++t) {
        size_t base = (size_t)(qbase + l16) * HD + t * 32 + hi * 8;
        qa[t].u[0] = *reinterpret_cast<const uint4*>(&qp[base]);
        qa[t].u[1] = *reinterpret_cast<const uint4*>(&qp[base + 16]);
    }

    float mrow[8], lrow[8];
    v8f acc[4];
#pragma unroll
    for (int e = 0; e < 8; ++e) { mrow[e] = -1e30f; lrow[e] = 0.0f; }
#pragma unroll
    for (int f = 0; f < 4; ++f)
#pragma unroll
        for (int e = 0; e < 8; ++e) acc[f][e] = 0.0f;

    for (int kb = 0; kb < qbase + 16; kb += 32) {
        // ---- scores: K^T fragments straight from global (d contiguous per key)
        ABFrag k00, k01, k10, k11;
        {
            size_t rl = (size_t)(kb + l16) * HD;       // keys kb..kb+15
            size_t rh = (size_t)(kb + 16 + l16) * HD;  // keys kb+16..kb+31
            int dk = hi * 8;
            k00.u[0] = *reinterpret_cast<const uint4*>(&kp[rl + dk]);
            k00.u[1] = *reinterpret_cast<const uint4*>(&kp[rl + dk + 16]);
            k01.u[0] = *reinterpret_cast<const uint4*>(&kp[rl + 32 + dk]);
            k01.u[1] = *reinterpret_cast<const uint4*>(&kp[rl + 32 + dk + 16]);
            k10.u[0] = *reinterpret_cast<const uint4*>(&kp[rh + dk]);
            k10.u[1] = *reinterpret_cast<const uint4*>(&kp[rh + dk + 16]);
            k11.u[0] = *reinterpret_cast<const uint4*>(&kp[rh + 32 + dk]);
            k11.u[1] = *reinterpret_cast<const uint4*>(&kp[rh + 32 + dk + 16]);
        }
        v8f s0, s1;
#pragma unroll
        for (int e = 0; e < 8; ++e) { s0[e] = 0.0f; s1[e] = 0.0f; }
        s0 = __builtin_amdgcn_wmma_f32_16x16x32_bf16(false, qa[0].v, false, k00.v, (short)0, s0, false, false);
        s0 = __builtin_amdgcn_wmma_f32_16x16x32_bf16(false, qa[1].v, false, k01.v, (short)0, s0, false, false);
        s1 = __builtin_amdgcn_wmma_f32_16x16x32_bf16(false, qa[0].v, false, k10.v, (short)0, s1, false, false);
        s1 = __builtin_amdgcn_wmma_f32_16x16x32_bf16(false, qa[1].v, false, k11.v, (short)0, s1, false, false);

        // ---- causal mask + online softmax (row r = e + 8*hi, col = key)
        float p0[8], p1[8], alpha[8];
#pragma unroll
        for (int e = 0; e < 8; ++e) {
            int row = qbase + e + hi * 8;
            float v0 = s0[e] * SCALE_ATT + ((kb + l16)      > row ? NEG_MASK : 0.0f);
            float v1 = s1[e] * SCALE_ATT + ((kb + 16 + l16) > row ? NEG_MASK : 0.0f);
            float bm   = rowmax16(fmaxf(v0, v1));
            float mnew = fmaxf(mrow[e], bm);
            float al   = __expf(mrow[e] - mnew);
            float e0   = __expf(v0 - mnew);
            float e1   = __expf(v1 - mnew);
            lrow[e] = lrow[e] * al + rowsum16(e0 + e1);
            mrow[e] = mnew;
            alpha[e] = al;
            p0[e] = e0; p1[e] = e1;
        }
#pragma unroll
        for (int f = 0; f < 4; ++f)
#pragma unroll
            for (int e = 0; e < 8; ++e) acc[f][e] *= alpha[e];

        // ---- P: C-layout -> A-layout via per-wave LDS tile (in-order DS per wave)
        unsigned short* pw = &Ps[wid][0];
#pragma unroll
        for (int e = 0; e < 8; ++e) {
            int r = e + hi * 8;
            pw[r * 40 + l16]      = f2bf(p0[e]);
            pw[r * 40 + 16 + l16] = f2bf(p1[e]);
        }
        ABFrag pa;
        {
            int c = hi * 8;
            pa.u[0] = *reinterpret_cast<const uint4*>(&pw[l16 * 40 + c]);
            pa.u[1] = *reinterpret_cast<const uint4*>(&pw[l16 * 40 + c + 16]);
        }

        // ---- PV: V^T fragments (s contiguous per d-row), 4 d-chunks of 16
#pragma unroll
        for (int f = 0; f < 4; ++f) {
            ABFrag vb;
            int d = f * 16 + l16;
            const unsigned short* vrow = &vp[(size_t)d * S_LEN + kb + hi * 8];
            vb.u[0] = *reinterpret_cast<const uint4*>(vrow);
            vb.u[1] = *reinterpret_cast<const uint4*>(vrow + 16);
            acc[f] = __builtin_amdgcn_wmma_f32_16x16x32_bf16(
                false, pa.v, false, vb.v, (short)0, acc[f], false, false);
        }
    }

    // ---- normalize + store ctx as bf16, layout [b][s][h*64+d] for out-proj GEMM
#pragma unroll
    for (int f = 0; f < 4; ++f)
#pragma unroll
        for (int e = 0; e < 8; ++e) {
            int row = qbase + e + hi * 8;
            int d   = f * 16 + l16;
            float o = acc[f][e] / lrow[e];
            ctxws[((size_t)b * S_LEN + row) * D_MODEL + h * HD + d] = f2bf(o);
        }
}

// ---------------------------------------------------------------------------
// Kernel 3: output projection  out = ctx @ Wo   (4096 x 1024, fp32 out)
// ---------------------------------------------------------------------------
__global__ __launch_bounds__(256)
void out_gemm_kernel(const unsigned short* __restrict__ Ag,   // bf16 ctx (4096x1024)
                     const unsigned short* __restrict__ Bg,   // bf16 WoT (1024x1024)
                     float* __restrict__ out) {
    __shared__ alignas(16) unsigned short As[128 * 40];
    __shared__ alignas(16) unsigned short Bs[128 * 40];

    const int tid  = threadIdx.x;
    const int wid  = tid >> 5;
    const int lane = tid & 31;
    const int l16  = lane & 15;
    const int hi   = (lane >> 4) & 1;
    const int waveM = wid & 1;
    const int waveN = wid >> 1;
    const int m0 = blockIdx.y * 128;
    const int n0 = blockIdx.x * 128;

    v8f acc[4][2];
#pragma unroll
    for (int a = 0; a < 4; ++a)
#pragma unroll
        for (int b = 0; b < 2; ++b)
#pragma unroll
            for (int e = 0; e < 8; ++e) acc[a][b][e] = 0.0f;

    for (int k0 = 0; k0 < D_MODEL; k0 += 32) {
        __syncthreads();
        {
            int j = tid;
#pragma unroll
            for (int it = 0; it < 2; ++it, j += 256) {
                int row = j >> 2, cc = (j & 3) << 3;
                *reinterpret_cast<uint4*>(&As[row * 40 + cc]) =
                    *reinterpret_cast<const uint4*>(&Ag[(size_t)(m0 + row) * D_MODEL + k0 + cc]);
                *reinterpret_cast<uint4*>(&Bs[row * 40 + cc]) =
                    *reinterpret_cast<const uint4*>(&Bg[(size_t)(n0 + row) * D_MODEL + k0 + cc]);
            }
        }
        if (k0 + 32 < D_MODEL) {
            __builtin_prefetch(&Ag[(size_t)(m0 + (tid & 127)) * D_MODEL + k0 + 32], 0, 3);
            __builtin_prefetch(&Bg[(size_t)(n0 + (tid & 127)) * D_MODEL + k0 + 32], 0, 3);
        }
        __syncthreads();

        ABFrag a[4], b[2];
#pragma unroll
        for (int fm = 0; fm < 4; ++fm) {
            int am = waveM * 64 + fm * 16 + l16;
            int c  = hi * 8;
            a[fm].u[0] = *reinterpret_cast<const uint4*>(&As[am * 40 + c]);
            a[fm].u[1] = *reinterpret_cast<const uint4*>(&As[am * 40 + c + 16]);
        }
#pragma unroll
        for (int fn = 0; fn < 2; ++fn) {
            int bn = waveN * 32 + fn * 16 + l16;
            int c  = hi * 8;
            b[fn].u[0] = *reinterpret_cast<const uint4*>(&Bs[bn * 40 + c]);
            b[fn].u[1] = *reinterpret_cast<const uint4*>(&Bs[bn * 40 + c + 16]);
        }
#pragma unroll
        for (int fm = 0; fm < 4; ++fm)
#pragma unroll
            for (int fn = 0; fn < 2; ++fn)
                acc[fm][fn] = __builtin_amdgcn_wmma_f32_16x16x32_bf16(
                    false, a[fm].v, false, b[fn].v, (short)0, acc[fm][fn], false, false);
    }

#pragma unroll
    for (int fm = 0; fm < 4; ++fm)
#pragma unroll
        for (int fn = 0; fn < 2; ++fn) {
            int n = n0 + waveN * 32 + fn * 16 + l16;
#pragma unroll
            for (int e = 0; e < 8; ++e) {
                int row = m0 + waveM * 64 + fm * 16 + e + hi * 8;
                out[(size_t)row * D_MODEL + n] = acc[fm][fn][e];
            }
        }
}

// ---------------------------------------------------------------------------
// Launch: prep (bf16 convert / weight transpose) -> QKV+RoPE -> attention ->
// out-proj. Workspace layout (bytes): hsb 8M | WqkvT 3M | WoT 2M | q 8M |
// k 2M | vT 2M | ctx 8M  = 33 MB total.
// ---------------------------------------------------------------------------
extern "C" void kernel_launch(void* const* d_in, const int* in_sizes, int n_in,
                              void* d_out, int out_size, void* d_ws, size_t ws_size,
                              hipStream_t stream) {
    const float* hs   = (const float*)d_in[0];
    const float* cosb = (const float*)d_in[1];
    const float* sinb = (const float*)d_in[2];
    // d_in[3] = attention_mask (pure causal; applied analytically in-kernel)
    const float* Wq = (const float*)d_in[4];
    const float* Wk = (const float*)d_in[5];
    const float* Wv = (const float*)d_in[6];
    const float* Wo = (const float*)d_in[7];
    float* out = (float*)d_out;

    char* ws = (char*)d_ws;
    unsigned short* hsb   = (unsigned short*)(ws);
    unsigned short* wqkvt = (unsigned short*)(ws + ((size_t)8  << 20));
    unsigned short* wot   = (unsigned short*)(ws + ((size_t)11 << 20));
    unsigned short* qws   = (unsigned short*)(ws + ((size_t)13 << 20));
    unsigned short* kws   = (unsigned short*)(ws + ((size_t)21 << 20));
    unsigned short* vtws  = (unsigned short*)(ws + ((size_t)23 << 20));
    unsigned short* ctxws = (unsigned short*)(ws + ((size_t)25 << 20));

    const int nHS = 2 * S_LEN * D_MODEL;                 // 4,194,304
    cvt_bf16_kernel<<<(nHS + 255) / 256, 256, 0, stream>>>(hs, hsb, nHS);
    pack_wqkv_kernel<<<(NQKV * D_MODEL + 255) / 256, 256, 0, stream>>>(Wq, Wk, Wv, wqkvt);
    transpose_wo_kernel<<<(D_MODEL * D_MODEL + 255) / 256, 256, 0, stream>>>(Wo, wot);

    qkv_gemm_kernel<<<dim3(NQKV / 128, 4096 / 128), 256, 0, stream>>>(
        hsb, wqkvt, cosb, sinb, qws, kws, vtws);

    attn_kernel<<<dim3(S_LEN / 128, N_H, 2), 256, 0, stream>>>(qws, kws, vtws, ctxws);

    out_gemm_kernel<<<dim3(D_MODEL / 128, 4096 / 128), 256, 0, stream>>>(ctxws, wot, out);
}
